// NTXentLoss_38010460569972
// MI455X (gfx1250) — matrix operand order repaired and checked by reading, not compiled, not run on hardware
//
#include <hip/hip_runtime.h>

typedef __attribute__((ext_vector_type(16))) _Float16 v16h;
typedef __attribute__((ext_vector_type(8)))  _Float16 v8h;
typedef __attribute__((ext_vector_type(8)))  float    v8f;
typedef __attribute__((ext_vector_type(4)))  float    v4f;

#define D_DIM 256
#define INV_T 2.0f   // 1 / 0.5

// ---------------------------------------------------------------------------
// Kernel 0: zero the rowsum array and the positive-pair accumulator.
// ---------------------------------------------------------------------------
__global__ __launch_bounds__(256) void ntx_zero(float* __restrict__ rowsum,
                                                float* __restrict__ posAcc,
                                                int n2) {
  int t = blockIdx.x * 256 + threadIdx.x;
  if (t < n2) rowsum[t] = 0.0f;
  if (t == 0) *posAcc = 0.0f;
}

// ---------------------------------------------------------------------------
// Kernel 1: per-row L2 normalize z_i / z_j -> f16 reps [2N x 256]; accumulate
// sum_i dot(zi_n, zj_n) (positive logits) in fp32; and emit per-row
// diagexp[r] = exp(selfdot_f16 / T) so the GEMM loop needs no diag masking.
// ---------------------------------------------------------------------------
__global__ __launch_bounds__(256) void ntx_normalize(
    const float* __restrict__ zi, const float* __restrict__ zj,
    _Float16* __restrict__ reps, float* __restrict__ diagexp,
    float* __restrict__ posAcc, int N) {
  int wave = threadIdx.x >> 5;
  int lane = threadIdx.x & 31;
  int i = blockIdx.x * 8 + wave;
  if (i >= N) return;

  const float* pi = zi + (size_t)i * D_DIM + lane * 8;
  const float* pj = zj + (size_t)i * D_DIM + lane * 8;
  v4f a0 = *(const v4f*)(pi);
  v4f a1 = *(const v4f*)(pi + 4);
  v4f b0 = *(const v4f*)(pj);
  v4f b1 = *(const v4f*)(pj + 4);

  float ssi = 0.f, ssj = 0.f, dot = 0.f;
#pragma unroll
  for (int e = 0; e < 4; ++e) {
    ssi += a0[e] * a0[e] + a1[e] * a1[e];
    ssj += b0[e] * b0[e] + b1[e] * b1[e];
    dot += a0[e] * b0[e] + a1[e] * b1[e];
  }
#pragma unroll
  for (int m = 16; m >= 1; m >>= 1) {
    ssi += __shfl_xor(ssi, m, 32);
    ssj += __shfl_xor(ssj, m, 32);
    dot += __shfl_xor(dot, m, 32);
  }
  float ri = 1.0f / fmaxf(sqrtf(ssi), 1e-12f);
  float rj = 1.0f / fmaxf(sqrtf(ssj), 1e-12f);

  v8h hi, hj;
#pragma unroll
  for (int e = 0; e < 4; ++e) {
    hi[e]     = (_Float16)(a0[e] * ri);
    hi[e + 4] = (_Float16)(a1[e] * ri);
    hj[e]     = (_Float16)(b0[e] * rj);
    hj[e + 4] = (_Float16)(b1[e] * rj);
  }
  *(v8h*)(reps + (size_t)i * D_DIM + lane * 8)       = hi;
  *(v8h*)(reps + (size_t)(i + N) * D_DIM + lane * 8) = hj;

  // Self-dot of the QUANTIZED rows (what the f16 WMMA will produce on the
  // diagonal), so finalize can subtract exp(selfdot/T) from each row sum.
  float di = 0.f, dj = 0.f;
#pragma unroll
  for (int e = 0; e < 8; ++e) {
    float fi = (float)hi[e], fj = (float)hj[e];
    di += fi * fi;
    dj += fj * fj;
  }
#pragma unroll
  for (int m = 16; m >= 1; m >>= 1) {
    di += __shfl_xor(di, m, 32);
    dj += __shfl_xor(dj, m, 32);
  }
  if (lane == 0) {
    diagexp[i]     = __expf(di * INV_T);
    diagexp[i + N] = __expf(dj * INV_T);
    atomicAdd(posAcc, dot * ri * rj);
  }
}

// ---------------------------------------------------------------------------
// Kernel 2 helpers
// ---------------------------------------------------------------------------
__device__ __forceinline__ void loadBFrags(v16h b[8], const _Float16* brow) {
#pragma unroll
  for (int kk = 0; kk < 8; ++kk)
    b[kk] = *(const v16h*)(brow + kk * 32);
}

__device__ __forceinline__ void tileAccum(const v16h aF0[8], const v16h aF1[8],
                                          const v16h b[8],
                                          float acc0[8], float acc1[8]) {
  v8f c0 = {};
  v8f c1 = {};
#pragma unroll
  for (int kk = 0; kk < 8; ++kk) {
    c0 = __builtin_amdgcn_wmma_f32_16x16x32_f16(
        false, aF0[kk], false, b[kk], (short)0, c0, false, false);
    c1 = __builtin_amdgcn_wmma_f32_16x16x32_f16(
        false, aF1[kk], false, b[kk], (short)0, c1, false, false);
  }
#pragma unroll
  for (int j = 0; j < 8; ++j) {
    acc0[j] += __expf(c0[j] * INV_T);
    acc1[j] += __expf(c1[j] * INV_T);
  }
}

// ---------------------------------------------------------------------------
// Kernel 2: fused sim = reps @ reps^T tile GEMM + exp(sim/T) accumulation.
// 32 rows per wave (two 16-row tiles share each B fragment). Double-buffered
// B fragments across column blocks so global loads overlap the WMMA chain.
// No diag masking here (handled via diagexp in finalize).
// ---------------------------------------------------------------------------
__global__ __launch_bounds__(256) void ntx_simexp(
    const _Float16* __restrict__ reps, float* __restrict__ rowsum,
    int cbPerStrip) {
  int wave  = threadIdx.x >> 5;
  int lane  = threadIdx.x & 31;
  int laneM = lane & 15;   // row (A) / col (B) within tile
  int laneH = lane >> 4;   // K-half selector

  int rowBase = (blockIdx.x * 8 + wave) * 32;   // 32 rows per wave

  // A fragments: lane holds one row; 16 contiguous K values starting at
  // K = kb + laneH*16 (matches B fragment order).
  v16h aFrag0[8], aFrag1[8];
  const _Float16* arow0 = reps + (size_t)(rowBase + laneM) * D_DIM;
  const _Float16* arow1 = reps + (size_t)(rowBase + 16 + laneM) * D_DIM;
#pragma unroll
  for (int kk = 0; kk < 8; ++kk) {
    int kb = kk * 32;
    v8h lo0 = *(const v8h*)(arow0 + kb + laneH * 16);
    v8h hi0 = *(const v8h*)(arow0 + kb + laneH * 16 + 8);
    v8h lo1 = *(const v8h*)(arow1 + kb + laneH * 16);
    v8h hi1 = *(const v8h*)(arow1 + kb + laneH * 16 + 8);
#pragma unroll
    for (int e = 0; e < 8; ++e) {
      aFrag0[kk][e] = lo0[e]; aFrag0[kk][e + 8] = hi0[e];
      aFrag1[kk][e] = lo1[e]; aFrag1[kk][e + 8] = hi1[e];
    }
  }

  float acc0[8] = {0.f, 0.f, 0.f, 0.f, 0.f, 0.f, 0.f, 0.f};
  float acc1[8] = {0.f, 0.f, 0.f, 0.f, 0.f, 0.f, 0.f, 0.f};

  int cb0 = blockIdx.y * cbPerStrip;
  int cb1 = cb0 + cbPerStrip;               // cbPerStrip is even
  // B fragment: lane holds column (colBase+laneM) = a row of reps; 16
  // contiguous K values starting at laneH*16 (+ kk*32 per fragment).
  const _Float16* bbase = reps + (size_t)laneM * D_DIM + laneH * 16;

  v16h bA[8], bB[8];
  loadBFrags(bA, bbase + (size_t)cb0 * 16 * D_DIM);

  for (int cb = cb0; cb < cb1; cb += 2) {
    // Loads for the NEXT tile fill while the current WMMA chain runs.
    loadBFrags(bB, bbase + (size_t)(cb + 1) * 16 * D_DIM);
    tileAccum(aFrag0, aFrag1, bA, acc0, acc1);

    int nx = (cb + 2 < cb1) ? cb + 2 : cb0;  // tail: harmless redundant load
    __builtin_prefetch(bbase + (size_t)nx * 16 * D_DIM, 0, 3);
    loadBFrags(bA, bbase + (size_t)nx * 16 * D_DIM);
    tileAccum(aFrag0, aFrag1, bB, acc0, acc1);
  }

  // Reduce each row's partial across its 16 owning lanes (within half-wave).
#pragma unroll
  for (int j = 0; j < 8; ++j) {
    float s0 = acc0[j];
    float s1 = acc1[j];
#pragma unroll
    for (int m = 8; m >= 1; m >>= 1) {
      s0 += __shfl_xor(s0, m, 32);
      s1 += __shfl_xor(s1, m, 32);
    }
    if (laneM == 0) {
      atomicAdd(&rowsum[rowBase + j + 8 * laneH], s0);
      atomicAdd(&rowsum[rowBase + 16 + j + 8 * laneH], s1);
    }
  }
}

// ---------------------------------------------------------------------------
// Kernel 3: loss = mean_r log(rowsum[r] - diagexp[r]) - (2/T) * posSum / (2N)
// ---------------------------------------------------------------------------
__global__ __launch_bounds__(256) void ntx_finalize(
    const float* __restrict__ rowsum, const float* __restrict__ diagexp,
    const float* __restrict__ posAcc, float* __restrict__ out, int n2) {
  __shared__ float red[256];
  float local = 0.0f;
  for (int r = threadIdx.x; r < n2; r += 256)
    local += __logf(rowsum[r] - diagexp[r]);
  red[threadIdx.x] = local;
  __syncthreads();
  for (int s = 128; s > 0; s >>= 1) {
    if (threadIdx.x < s) red[threadIdx.x] += red[threadIdx.x + s];
    __syncthreads();
  }
  if (threadIdx.x == 0) {
    float lseSum = red[0];
    float posSum = *posAcc;  // sum over i of pos_i (each appears twice in 2N rows)
    out[0] = lseSum / (float)n2 - (posSum * 2.0f * INV_T) / (float)n2;
  }
}

// ---------------------------------------------------------------------------
extern "C" void kernel_launch(void* const* d_in, const int* in_sizes, int n_in,
                              void* d_out, int out_size, void* d_ws, size_t ws_size,
                              hipStream_t stream) {
  const float* zi = (const float*)d_in[0];
  const float* zj = (const float*)d_in[1];
  int N  = in_sizes[0] / D_DIM;  // 4096
  int n2 = 2 * N;                // 8192

  char* ws = (char*)d_ws;
  _Float16* reps = (_Float16*)ws;                       // [2N x 256] f16: 4 MB
  size_t repsBytes = (size_t)n2 * D_DIM * sizeof(_Float16);
  float* rowsum  = (float*)(ws + repsBytes);            // [2N] f32
  float* diagexp = rowsum + n2;                         // [2N] f32
  float* posAcc  = diagexp + n2;                        // 1 f32

  ntx_zero<<<(n2 + 255) / 256, 256, 0, stream>>>(rowsum, posAcc, n2);
  ntx_normalize<<<(N + 7) / 8, 256, 0, stream>>>(zi, zj, reps, diagexp, posAcc, N);

  int superBlocks = n2 / 32;                // 256 (32 rows per wave)
  int strips      = 16;
  int cbPerStrip  = (n2 / 16) / strips;     // 32 column blocks per strip (even)
  dim3 grid(superBlocks / 8, strips);       // 32 x 16 = 512 blocks, 4096 waves
  ntx_simexp<<<grid, 256, 0, stream>>>(reps, rowsum, cbPerStrip);

  ntx_finalize<<<1, 256, 0, stream>>>(rowsum, diagexp, posAcc, (float*)d_out, n2);
}
